// XPredNextScale_84052509983438
// MI455X (gfx1250) — matrix operand order repaired and checked by reading, not compiled, not run on hardware
//
#include <hip/hip_runtime.h>

#define B_      8
#define L_      341
#define D_      1024
#define H_      16
#define HD_     64
#define MLP_    4096
#define DEPTH_  8
#define PD_     192
#define PATCHL_ 340
#define EPSLN   1e-6f

typedef float v8f   __attribute__((ext_vector_type(8)));
typedef float fvec4 __attribute__((ext_vector_type(4)));
typedef __bf16 bf16;
typedef bf16  v2bf  __attribute__((ext_vector_type(2)));
typedef bf16  v4bf  __attribute__((ext_vector_type(4)));
typedef bf16  v16bf __attribute__((ext_vector_type(16)));

union BFrag { v16bf v; v2bf h[8]; };    // fill a v16bf pairwise (no &vec[i])

__device__ __forceinline__ bf16 f2bf(float f) { return (bf16)f; }

__device__ __forceinline__ v2bf pk2(float lo, float hi) {
#if __has_builtin(__builtin_amdgcn_cvt_pk_bf16_f32)
  return __builtin_amdgcn_cvt_pk_bf16_f32(lo, hi);
#else
  v2bf r; r[0] = (bf16)lo; r[1] = (bf16)hi; return r;
#endif
}

__device__ __forceinline__ float gelu_tanh(float x) {
  float x3 = x * x * x;
  return 0.5f * x * (1.0f + tanhf(0.7978845608028654f * (x + 0.044715f * x3)));
}

// -------------------------------------------------------------------------
// bf16-WMMA GEMM: C[MxN] = epi(A[MxK] * B[KxN] + bias)
// epi: 0 bias, 1 bias+gelu, 2 res + gate*(acc+bias)  (adaLN gate, per b = m/L)
// K must be a multiple of 32 (true for all call sites: 192/1024/4096).
// 256 threads = 8 waves (4x2), tile 128x128, wave tile 32x64, 8 wmma/k-step.
// Uniform interior fast path: no per-vector exec-masking in the steady state.
// -------------------------------------------------------------------------
__global__ __launch_bounds__(256)
void gemm_wmma_bf16(const float* __restrict__ A, const float* __restrict__ Bm,
                    float* __restrict__ C, int M, int N, int K,
                    const float* __restrict__ bias,
                    const float* __restrict__ gate, int gate_stride, int rows_per_b,
                    const float* __restrict__ res, int epi)
{
  __shared__ __align__(16) bf16 lA[128][40];   // [m][k], padded
  __shared__ __align__(16) bf16 lB[128][40];   // [n][k] transposed, padded

  const int n0 = blockIdx.x * 128;
  const int m0 = blockIdx.y * 128;
  const int t  = threadIdx.x;
  const int w  = t >> 5, lane = t & 31;
  const int wm = w >> 1, wn = w & 1;           // 4x2 wave grid
  const int g  = lane >> 4, ln = lane & 15;

  const bool intA = (m0 + 128 <= M);           // block-uniform interior tests
  const bool intB = (n0 + 128 <= N);

  v8f acc[2][4];
  #pragma unroll
  for (int i = 0; i < 2; i++)
    #pragma unroll
    for (int j = 0; j < 4; j++)
      #pragma unroll
      for (int r = 0; r < 8; r++) acc[i][j][r] = 0.0f;

  const int bn4  = (t & 31) * 4;   // B-fill: this thread's n offset
  const int bkp  = t >> 5;         // B-fill: k-pair slot 0..7
  const int gn_b = n0 + bn4;

  for (int k0 = 0; k0 < K; k0 += 32) {
    // ---- stage A: 128x32, 4 float4 per thread ----
    if (intA) {
      #pragma unroll
      for (int j = 0; j < 4; j++) {
        const int linear = t + 256 * j;
        const int row = linear >> 3;
        const int c4  = (linear & 7) * 4;
        const fvec4 v = *(const fvec4*)(A + (size_t)(m0 + row) * K + k0 + c4);
        v4bf p; p.lo = pk2(v[0], v[1]); p.hi = pk2(v[2], v[3]);
        *(v4bf*)&lA[row][c4] = p;
      }
    } else {
      #pragma unroll
      for (int j = 0; j < 4; j++) {
        const int linear = t + 256 * j;
        const int row = linear >> 3;
        const int c4  = (linear & 7) * 4;
        const int gm  = m0 + row;
        fvec4 v = {0.f, 0.f, 0.f, 0.f};
        if (gm < M) v = *(const fvec4*)(A + (size_t)gm * K + k0 + c4);
        v4bf p; p.lo = pk2(v[0], v[1]); p.hi = pk2(v[2], v[3]);
        *(v4bf*)&lA[row][c4] = p;
      }
    }
    // ---- stage B: 32x128 transposed; 2 k-rows per 32-bit LDS store ----
    if (intB) {
      #pragma unroll
      for (int j = 0; j < 2; j++) {
        const int k2 = (bkp + 8 * j) * 2;
        const fvec4 r0 = *(const fvec4*)(Bm + (size_t)(k0 + k2) * N + gn_b);
        const fvec4 r1 = *(const fvec4*)(Bm + (size_t)(k0 + k2 + 1) * N + gn_b);
        #pragma unroll
        for (int i = 0; i < 4; i++)
          *(v2bf*)&lB[bn4 + i][k2] = pk2(r0[i], r1[i]);
      }
      if (k0 + 32 < K)
        __builtin_prefetch(Bm + (size_t)(k0 + 32) * N + gn_b, 0, 1);
    } else {
      #pragma unroll
      for (int j = 0; j < 2; j++) {
        const int k2 = (bkp + 8 * j) * 2;
        fvec4 r0 = {0.f, 0.f, 0.f, 0.f}, r1 = {0.f, 0.f, 0.f, 0.f};
        if (gn_b < N) {
          r0 = *(const fvec4*)(Bm + (size_t)(k0 + k2) * N + gn_b);
          r1 = *(const fvec4*)(Bm + (size_t)(k0 + k2 + 1) * N + gn_b);
        }
        #pragma unroll
        for (int i = 0; i < 4; i++)
          *(v2bf*)&lB[bn4 + i][k2] = pk2(r0[i], r1[i]);
      }
    }
    __syncthreads();

    // ---- fragments + 8 WMMAs ----
    v16bf af[2], bfr[4];
    #pragma unroll
    for (int fm = 0; fm < 2; fm++) {
      const int row = wm * 32 + fm * 16 + ln;
      #pragma unroll
      for (int e = 0; e < 16; e++) {
        const int kk = 16 * g + e;                     // contiguous K pattern (matches B)
        af[fm][e] = lA[row][kk];
      }
    }
    #pragma unroll
    for (int fn = 0; fn < 4; fn++) {
      const int col = wn * 64 + fn * 16 + ln;
      #pragma unroll
      for (int e = 0; e < 16; e++)
        bfr[fn][e] = lB[col][16 * g + e];              // contiguous K pattern (matches A)
    }
    #pragma unroll
    for (int fm = 0; fm < 2; fm++)
      #pragma unroll
      for (int fn = 0; fn < 4; fn++)
        acc[fm][fn] = __builtin_amdgcn_wmma_f32_16x16x32_bf16(
            false, af[fm], false, bfr[fn], (short)0, acc[fm][fn], false, false);
    __syncthreads();
  }

  // ---- epilogue (C/D layout: M = r + 8*g, N = ln) ----
  #pragma unroll
  for (int fm = 0; fm < 2; fm++)
    #pragma unroll
    for (int fn = 0; fn < 4; fn++)
      #pragma unroll
      for (int r = 0; r < 8; r++) {
        const int m = m0 + wm * 32 + fm * 16 + 8 * g + r;
        const int n = n0 + wn * 64 + fn * 16 + ln;
        if ((intA && intB) || (m < M && n < N)) {
          float v = acc[fm][fn][r];
          if (bias) v += bias[n];
          if (epi == 1) {
            v = gelu_tanh(v);
          } else if (epi == 2) {
            const int b = m / rows_per_b;
            v = res[(size_t)m * N + n] + gate[(size_t)b * gate_stride + n] * v;
          }
          C[(size_t)m * N + n] = v;
        }
      }
}

// -------------------------------------------------------------------------
__global__ __launch_bounds__(256)
void cond_kernel(const float* __restrict__ class_emb, const int* __restrict__ labels,
                 float* __restrict__ cond, float* __restrict__ scond)
{
  const int b = blockIdx.x;
  const int lab = labels[b];
  for (int j = 0; j < 4; j++) {
    const int d = threadIdx.x + 256 * j;
    const float v = class_emb[(size_t)lab * D_ + d];
    cond[b * D_ + d]  = v;
    scond[b * D_ + d] = v / (1.0f + __expf(-v));
  }
}

__global__ __launch_bounds__(256)
void embed_kernel(const float* __restrict__ tok, const float* __restrict__ cond,
                  const float* __restrict__ pos, const float* __restrict__ scale_emb,
                  float* __restrict__ x)
{
  const int row = blockIdx.x;       // b*L + l
  const int b = row / L_, l = row % L_;
  const int sid = (l < 5) ? 0 : (l < 21) ? 1 : (l < 85) ? 2 : 3;
  for (int j = 0; j < 4; j++) {
    const int d = threadIdx.x + 256 * j;
    const float base = (l == 0) ? cond[b * D_ + d]
                                : tok[((size_t)b * PATCHL_ + (l - 1)) * D_ + d];
    x[(size_t)row * D_ + d] = base + pos[(size_t)l * D_ + d]
                            + scale_emb[(size_t)sid * D_ + d];
  }
}

__global__ __launch_bounds__(256)
void ln_mod_kernel(const float* __restrict__ x, float* __restrict__ out,
                   const float* __restrict__ sgate, const float* __restrict__ shift,
                   int srow_stride)
{
  const int row = blockIdx.x;
  const int b = row / L_;
  const float* xr = x + (size_t)row * D_;
  float lv[4], sum = 0.f, sq = 0.f;
  for (int j = 0; j < 4; j++) {
    const float v = xr[threadIdx.x + 256 * j];
    lv[j] = v; sum += v; sq += v * v;
  }
  __shared__ float rs[256], rq[256];
  rs[threadIdx.x] = sum; rq[threadIdx.x] = sq;
  __syncthreads();
  for (int o = 128; o > 0; o >>= 1) {
    if (threadIdx.x < o) { rs[threadIdx.x] += rs[threadIdx.x + o];
                           rq[threadIdx.x] += rq[threadIdx.x + o]; }
    __syncthreads();
  }
  const float mean = rs[0] * (1.0f / D_);
  const float var  = rq[0] * (1.0f / D_) - mean * mean;
  const float inv  = rsqrtf(var + EPSLN);
  const float* sg = sgate + (size_t)b * srow_stride;
  const float* sh = shift + (size_t)b * srow_stride;
  for (int j = 0; j < 4; j++) {
    const int d = threadIdx.x + 256 * j;
    out[(size_t)row * D_ + d] = (lv[j] - mean) * inv * (1.0f + sg[d]) + sh[d];
  }
}

__global__ __launch_bounds__(32)
void qkv_post_kernel(const float* __restrict__ qkv, const float* __restrict__ qscale,
                     float* __restrict__ qf, float* __restrict__ kf, float* __restrict__ vf)
{
  const int idx = blockIdx.x;              // (b*L + l)*H + h
  const int h = idx % H_;
  const int bl = idx / H_;
  const int l = bl % L_, b = bl / L_;
  const int lane = threadIdx.x;
  const float* base = qkv + (size_t)bl * 3 * D_ + h * HD_;
  const float q0 = base[lane],          q1 = base[lane + 32];
  const float k0 = base[D_ + lane],     k1 = base[D_ + lane + 32];
  const float v0 = base[2 * D_ + lane], v1 = base[2 * D_ + lane + 32];
  float sq = q0 * q0 + q1 * q1;
  float sk = k0 * k0 + k1 * k1;
  for (int o = 16; o > 0; o >>= 1) { sq += __shfl_xor(sq, o); sk += __shfl_xor(sk, o); }
  const float qn = __expf(qscale[h]) / fmaxf(sqrtf(sq), 1e-12f);
  const float kn = 1.0f / fmaxf(sqrtf(sk), 1e-12f);
  const size_t dst = ((size_t)(b * H_ + h) * L_ + l) * HD_;
  qf[dst + lane] = q0 * qn; qf[dst + lane + 32] = q1 * qn;
  kf[dst + lane] = k0 * kn; kf[dst + lane + 32] = k1 * kn;
  vf[dst + lane] = v0;      vf[dst + lane + 32] = v1;
}

// -------------------------------------------------------------------------
// Block-causal attention, one wave per (b, h, 16-query tile).
// Scores via WMMA (q frags hoisted, float4 loads), softmax in LDS with
// per-row key limits, then P*V via WMMA with V staged through LDS.
// -------------------------------------------------------------------------
__global__ __launch_bounds__(32)
void attn_kernel(const float* __restrict__ qf, const float* __restrict__ kf,
                 const float* __restrict__ vf, float* __restrict__ obuf)
{
  __shared__ float s[16][352];
  __shared__ __align__(16) float vt[32][68];
  const int qt = blockIdx.x, h = blockIdx.y, b = blockIdx.z;
  const int q0 = qt * 16;
  const int lane = threadIdx.x, g = lane >> 4, ln = lane & 15;
  const float* qb = qf + (size_t)(b * H_ + h) * L_ * HD_;
  const float* kb = kf + (size_t)(b * H_ + h) * L_ * HD_;
  const float* vb = vf + (size_t)(b * H_ + h) * L_ * HD_;

  const int qlast  = (q0 + 15 < L_ - 1) ? q0 + 15 : L_ - 1;
  const int maxlim = (qlast < 5) ? 5 : (qlast < 21) ? 21 : (qlast < 85) ? 85 : L_;
  const int ktiles = (maxlim + 15) >> 4;

  // hoisted q fragments (loop-invariant across key tiles), float4 loads
  BFrag aq[2];
  {
    const int row = q0 + ln;
    if (row < L_) {
      const fvec4* qr = (const fvec4*)(qb + (size_t)row * HD_);
      #pragma unroll
      for (int ks = 0; ks < 2; ks++) {
        const int i0 = 8 * ks + 2 * g;
        const fvec4 c0 = qr[i0], c1 = qr[i0 + 1], c2 = qr[i0 + 4], c3 = qr[i0 + 5];
        aq[ks].h[0] = pk2(c0[0], c0[1]); aq[ks].h[1] = pk2(c0[2], c0[3]);
        aq[ks].h[2] = pk2(c1[0], c1[1]); aq[ks].h[3] = pk2(c1[2], c1[3]);
        aq[ks].h[4] = pk2(c2[0], c2[1]); aq[ks].h[5] = pk2(c2[2], c2[3]);
        aq[ks].h[6] = pk2(c3[0], c3[1]); aq[ks].h[7] = pk2(c3[2], c3[3]);
      }
    } else {
      #pragma unroll
      for (int ks = 0; ks < 2; ks++)
        #pragma unroll
        for (int e = 0; e < 8; e++) aq[ks].h[e] = pk2(0.0f, 0.0f);
    }
  }

  // phase 1: scores
  for (int kt = 0; kt < ktiles; kt++) {
    v8f sa;
    #pragma unroll
    for (int r = 0; r < 8; r++) sa[r] = 0.0f;
    const int key = kt * 16 + ln;
    #pragma unroll
    for (int ks = 0; ks < 2; ks++) {
      BFrag bb;
      if (key < L_) {
        const fvec4* kr = (const fvec4*)(kb + (size_t)key * HD_);
        const int i0 = 8 * ks + 2 * g;
        const fvec4 c0 = kr[i0], c1 = kr[i0 + 1], c2 = kr[i0 + 4], c3 = kr[i0 + 5];
        bb.h[0] = pk2(c0[0], c0[1]); bb.h[1] = pk2(c0[2], c0[3]);
        bb.h[2] = pk2(c1[0], c1[1]); bb.h[3] = pk2(c1[2], c1[3]);
        bb.h[4] = pk2(c2[0], c2[1]); bb.h[5] = pk2(c2[2], c2[3]);
        bb.h[6] = pk2(c3[0], c3[1]); bb.h[7] = pk2(c3[2], c3[3]);
      } else {
        #pragma unroll
        for (int e = 0; e < 8; e++) bb.h[e] = pk2(0.0f, 0.0f);
      }
      sa = __builtin_amdgcn_wmma_f32_16x16x32_bf16(false, aq[ks].v, false, bb.v,
                                                   (short)0, sa, false, false);
    }
    #pragma unroll
    for (int r = 0; r < 8; r++) s[r + 8 * g][kt * 16 + ln] = sa[r];
  }
  __syncthreads();

  // phase 2: per-row softmax with key limit (== the -1e30 block mask)
  if (lane < 16) {
    const int row = q0 + lane;
    if (row < L_) {
      const int lim = (row < 5) ? 5 : (row < 21) ? 21 : (row < 85) ? 85 : L_;
      float mx = -1e30f;
      for (int k = 0; k < lim; k++) mx = fmaxf(mx, s[lane][k]);
      float den = 0.0f;
      for (int k = 0; k < lim; k++) { const float e = __expf(s[lane][k] - mx);
                                      s[lane][k] = e; den += e; }
      const float inv = 1.0f / den;
      for (int k = 0; k < lim; k++) s[lane][k] *= inv;
      for (int k = lim; k < 352; k++) s[lane][k] = 0.0f;
    } else {
      for (int k = 0; k < 352; k++) s[lane][k] = 0.0f;
    }
  }
  __syncthreads();

  // phase 3: O = P * V, V staged through LDS (coalesced b128 loads)
  v8f oacc[4];
  #pragma unroll
  for (int f = 0; f < 4; f++)
    #pragma unroll
    for (int r = 0; r < 8; r++) oacc[f][r] = 0.0f;
  const int k3 = (maxlim + 31) >> 5;
  for (int kt = 0; kt < k3; kt++) {
    for (int idx = lane; idx < 512; idx += 32) {      // 32 keys x 64 dims
      const int r = idx >> 4, c4 = (idx & 15) * 4;
      const int key = kt * 32 + r;
      fvec4 vv = {0.f, 0.f, 0.f, 0.f};
      if (key < L_) vv = *(const fvec4*)(vb + (size_t)key * HD_ + c4);
      *(fvec4*)&vt[r][c4] = vv;
    }
    __syncthreads();
    BFrag a;
    #pragma unroll
    for (int e = 0; e < 8; e++) {
      const int keyA = kt * 32 + 2 * e + 8 * g + (e >= 4 ? 8 : 0);
      a.h[e] = pk2(s[ln][keyA], s[ln][keyA + 1]);
    }
    #pragma unroll
    for (int fn = 0; fn < 4; fn++) {
      BFrag bb;
      const int d = fn * 16 + ln;
      #pragma unroll
      for (int e = 0; e < 8; e++)
        bb.h[e] = pk2(vt[2 * e + 8 * g + (e >= 4 ? 8 : 0)][d], vt[2 * e + 8 * g + (e >= 4 ? 8 : 0) + 1][d]);
      oacc[fn] = __builtin_amdgcn_wmma_f32_16x16x32_bf16(false, a.v, false, bb.v,
                                                         (short)0, oacc[fn], false, false);
    }
    __syncthreads();
  }

  #pragma unroll
  for (int fn = 0; fn < 4; fn++)
    #pragma unroll
    for (int r = 0; r < 8; r++) {
      const int m = q0 + r + 8 * g;
      if (m < L_)
        obuf[(size_t)(b * L_ + m) * D_ + h * HD_ + fn * 16 + ln] = oacc[fn][r];
    }
}

// -------------------------------------------------------------------------
extern "C" void kernel_launch(void* const* d_in, const int* in_sizes, int n_in,
                              void* d_out, int out_size, void* d_ws, size_t ws_size,
                              hipStream_t stream) {
  (void)in_sizes; (void)n_in; (void)out_size; (void)ws_size;
  const float* patches    = (const float*)d_in[0];
  const int*   labels     = (const int*)  d_in[1];
  const float* W_patch    = (const float*)d_in[2];
  const float* b_patch    = (const float*)d_in[3];
  const float* pos        = (const float*)d_in[4];
  const float* scale_emb  = (const float*)d_in[5];
  const float* class_emb  = (const float*)d_in[6];
  const float* ada_W      = (const float*)d_in[7];
  const float* ada_b      = (const float*)d_in[8];
  const float* qkv_W      = (const float*)d_in[9];
  const float* qkv_b      = (const float*)d_in[10];
  const float* q_scale    = (const float*)d_in[11];
  const float* proj_W     = (const float*)d_in[12];
  const float* proj_b     = (const float*)d_in[13];
  const float* fc1_W      = (const float*)d_in[14];
  const float* fc1_b      = (const float*)d_in[15];
  const float* fc2_W      = (const float*)d_in[16];
  const float* fc2_b      = (const float*)d_in[17];
  const float* head_ada_W = (const float*)d_in[18];
  const float* head_ada_b = (const float*)d_in[19];
  const float* head_W     = (const float*)d_in[20];
  const float* head_b     = (const float*)d_in[21];

  const int M = B_ * L_;  // 2728

  size_t off = 0;
  auto carve = [&](size_t elems) -> float* {
    float* p = (float*)((char*)d_ws + off);
    off += ((elems * sizeof(float)) + 255) & ~(size_t)255;
    return p;
  };
  float* tok   = carve((size_t)B_ * PATCHL_ * D_);
  float* x     = carve((size_t)M * D_);
  float* hbuf  = carve((size_t)M * D_);
  float* cond  = carve((size_t)B_ * D_);
  float* scond = carve((size_t)B_ * D_);
  float* ada   = carve((size_t)B_ * 6 * D_);
  float* qkv   = carve((size_t)M * 3 * D_);
  float* qfb   = carve((size_t)B_ * H_ * L_ * HD_);
  float* kfb   = carve((size_t)B_ * H_ * L_ * HD_);
  float* vfb   = carve((size_t)B_ * H_ * L_ * HD_);
  float* obuf  = carve((size_t)M * D_);
  float* mid   = carve((size_t)M * MLP_);
  float* adah  = carve((size_t)B_ * 2 * D_);

  auto gemm = [&](const float* A, const float* Bm, float* C, int m, int n, int k,
                  const float* bias, const float* gate, int gstride,
                  const float* res, int epi) {
    dim3 grid((n + 127) / 128, (m + 127) / 128);
    gemm_wmma_bf16<<<grid, dim3(256), 0, stream>>>(A, Bm, C, m, n, k, bias,
                                                   gate, gstride, L_, res, epi);
  };

  cond_kernel<<<B_, 256, 0, stream>>>(class_emb, labels, cond, scond);
  gemm(patches, W_patch, tok, B_ * PATCHL_, D_, PD_, b_patch, nullptr, 0, nullptr, 0);
  embed_kernel<<<M, 256, 0, stream>>>(tok, cond, pos, scale_emb, x);

  for (int i = 0; i < DEPTH_; i++) {
    const float* aW  = ada_W  + (size_t)i * D_ * 6 * D_;
    const float* ab  = ada_b  + (size_t)i * 6 * D_;
    const float* qW  = qkv_W  + (size_t)i * D_ * 3 * D_;
    const float* qb  = qkv_b  + (size_t)i * 3 * D_;
    const float* pW  = proj_W + (size_t)i * D_ * D_;
    const float* pb  = proj_b + (size_t)i * D_;
    const float* f1W = fc1_W  + (size_t)i * D_ * MLP_;
    const float* f1b = fc1_b  + (size_t)i * MLP_;
    const float* f2W = fc2_W  + (size_t)i * MLP_ * D_;
    const float* f2b = fc2_b  + (size_t)i * D_;

    gemm(scond, aW, ada, B_, 6 * D_, D_, ab, nullptr, 0, nullptr, 0);
    ln_mod_kernel<<<M, 256, 0, stream>>>(x, hbuf, ada + 2 * D_, ada + 4 * D_, 6 * D_);
    gemm(hbuf, qW, qkv, M, 3 * D_, D_, qb, nullptr, 0, nullptr, 0);
    qkv_post_kernel<<<M * H_, 32, 0, stream>>>(qkv, q_scale + (size_t)i * H_,
                                               qfb, kfb, vfb);
    attn_kernel<<<dim3((L_ + 15) / 16, H_, B_), 32, 0, stream>>>(qfb, kfb, vfb, obuf);
    gemm(obuf, pW, x, M, D_, D_, pb, ada /*g1*/, 6 * D_, x, 2);
    ln_mod_kernel<<<M, 256, 0, stream>>>(x, hbuf, ada + 3 * D_, ada + 5 * D_, 6 * D_);
    gemm(hbuf, f1W, mid, M, MLP_, D_, f1b, nullptr, 0, nullptr, 1);
    gemm(mid, f2W, x, M, D_, MLP_, f2b, ada + D_ /*g2*/, 6 * D_, x, 2);
  }

  gemm(scond, head_ada_W, adah, B_, 2 * D_, D_, head_ada_b, nullptr, 0, nullptr, 0);
  ln_mod_kernel<<<M, 256, 0, stream>>>(x, hbuf, adah, adah + D_, 2 * D_);
  gemm(hbuf, head_W, (float*)d_out, M, PD_, D_, head_b, nullptr, 0, nullptr, 0);
}
